// ChakraBlock_77275051589881
// MI455X (gfx1250) — compile-verified, matrix operand order checked by reading, not run on hardware
//
#include <hip/hip_runtime.h>
#include <hip/hip_bf16.h>
#include <math.h>
#include <stdint.h>

typedef __attribute__((ext_vector_type(8)))  __bf16 v8bf;
typedef __attribute__((ext_vector_type(16))) __bf16 v16bf;
typedef __attribute__((ext_vector_type(8)))  float  v8f;
typedef __attribute__((ext_vector_type(4)))  unsigned int v4u;
typedef __attribute__((ext_vector_type(4)))  int    v4i;
typedef __attribute__((ext_vector_type(8)))  int    v8i;

#define DMODEL 1024
#define TSEQ   2048
#define NBATCH 2
#define NHEAD  16
#define HDIM   64
#define MROWS  (NBATCH * TSEQ)   // 4096
#define DINNER 4096

// ---------------------------------------------------------------------------
// Fragment loader: per-lane 16 bf16 K-values as two contiguous 8-half chunks
// at k0 and k0+16 (k0 = (lane>=16 ? 8 : 0)), matching CDNA5 16-bit A/B layout.
// ---------------------------------------------------------------------------
__device__ __forceinline__ v16bf ld_frag(const __bf16* rowbase, int k0) {
    v8bf lo = *(const v8bf*)(rowbase + k0);
    v8bf hi = *(const v8bf*)(rowbase + k0 + 16);
    v16bf r;
#pragma unroll
    for (int i = 0; i < 8; ++i) { r[i] = lo[i]; r[i + 8] = hi[i]; }
    return r;
}

__device__ __forceinline__ v8f wmma_bf16(v16bf a, v16bf b, v8f c) {
    return __builtin_amdgcn_wmma_f32_16x16x32_bf16(false, a, false, b,
                                                   (short)0, c, false, false);
}

// ---------------------------------------------------------------------------
// TDM 2D tile load: rows x row_halfs bf16 tile, global row stride
// stride_elems, into LDS at lds_byte_off with HW row padding.
// pad_interval_code: dwords-before-pad = 2<<code ; pad_amount_code: pad dwords-1
// Issue from ONE wave only (TDM is wave-level; EXEC ignored).
// ---------------------------------------------------------------------------
__device__ __forceinline__ void tdm_load_2d(const void* gaddr, uint32_t lds_byte_off,
                                            uint32_t row_halfs, uint32_t rows,
                                            uint32_t stride_elems,
                                            uint32_t pad_interval_code,
                                            uint32_t pad_amount_code) {
    const uint64_t ga = (uint64_t)(uintptr_t)gaddr;
    v4u g0;
    g0[0] = 1u;                                    // count=1, user descriptor
    g0[1] = lds_byte_off;                          // lds_addr (bytes)
    g0[2] = (uint32_t)ga;                          // global_addr[31:0]
    g0[3] = (uint32_t)((ga >> 32) & 0x01FFFFFFu)   // global_addr[56:32]
          | (2u << 30);                            // type = 2 ("image")
    const uint32_t dim0 = stride_elems;            // tensor_dim0 (in-bounds tiles)
    const uint32_t dim1 = 0x40000000u;             // tensor_dim1 (never OOB)
    v8i g1;
    g1[0] = (int)((1u << 16)                       // data_size = 2 bytes
                | (1u << 20)                       // pad_enable
                | (pad_interval_code << 22)
                | (pad_amount_code << 25));
    g1[1] = (int)((dim0 & 0xFFFFu) << 16);         // [31:16]=tensor_dim0 lo
    g1[2] = (int)((dim0 >> 16) | ((dim1 & 0xFFFFu) << 16));
    g1[3] = (int)(((dim1 >> 16) & 0xFFFFu) | (row_halfs << 16)); // tile_dim0
    g1[4] = (int)(rows & 0xFFFFu);                 // tile_dim1 ; tile_dim2=0
    g1[5] = (int)stride_elems;                     // tensor_dim0_stride lo32
    g1[6] = 0;                                     // stride hi16 | dim1_stride lo
    g1[7] = 0;
    const v4i z4 = {0, 0, 0, 0};
#if __has_include(<hip/amd_detail/amd_gfx1250_TDM.h>)
    const v8i z8 = {0, 0, 0, 0, 0, 0, 0, 0};
    __builtin_amdgcn_tensor_load_to_lds(g0, g1, z4, z4, z8, 0);
#else
    __builtin_amdgcn_tensor_load_to_lds(g0, g1, z4, z4, 0);
#endif
}

// ---------------------------------------------------------------------------
// fp32 [K][N] -> bf16 [N][K] transposing convert (weights, once per launch)
// ---------------------------------------------------------------------------
__global__ __launch_bounds__(256) void cvt_transpose_bf16(const float* __restrict__ src,
                                                          __bf16* __restrict__ dst,
                                                          int K, int N) {
    __shared__ float tile[32][33];
    const int n0 = blockIdx.x * 32, k0 = blockIdx.y * 32;
    const int tx = threadIdx.x & 31, ty = threadIdx.x >> 5;  // ty in 0..7
#pragma unroll
    for (int i = 0; i < 4; ++i)
        tile[ty + 8 * i][tx] = src[(size_t)(k0 + ty + 8 * i) * N + n0 + tx];
    __syncthreads();
#pragma unroll
    for (int i = 0; i < 4; ++i)
        dst[(size_t)(n0 + ty + 8 * i) * K + k0 + tx] = (__bf16)tile[tx][ty + 8 * i];
}

// ---------------------------------------------------------------------------
// LayerNorm over D=1024, one block (256 threads) per row, bf16 output
// ---------------------------------------------------------------------------
__global__ __launch_bounds__(256) void ln_bf16_kernel(const float* __restrict__ x,
                                                      const float* __restrict__ g,
                                                      const float* __restrict__ b,
                                                      __bf16* __restrict__ out) {
    __shared__ float red[16];
    const int row = blockIdx.x;
    const float4 v = ((const float4*)(x + (size_t)row * DMODEL))[threadIdx.x];
    float s  = v.x + v.y + v.z + v.w;
    float s2 = v.x * v.x + v.y * v.y + v.z * v.z + v.w * v.w;
#pragma unroll
    for (int off = 16; off > 0; off >>= 1) {
        s  += __shfl_xor(s,  off, 32);
        s2 += __shfl_xor(s2, off, 32);
    }
    const int wave = threadIdx.x >> 5, lane = threadIdx.x & 31;
    if (lane == 0) { red[wave] = s; red[8 + wave] = s2; }
    __syncthreads();
    float S = 0.f, S2 = 0.f;
#pragma unroll
    for (int i = 0; i < 8; ++i) { S += red[i]; S2 += red[8 + i]; }
    const float mu  = S * (1.0f / DMODEL);
    const float var = S2 * (1.0f / DMODEL) - mu * mu;
    const float rs  = rsqrtf(var + 1e-5f);
    const float4 gg = ((const float4*)g)[threadIdx.x];
    const float4 bb = ((const float4*)b)[threadIdx.x];
    __bf16* o = out + (size_t)row * DMODEL + threadIdx.x * 4;
    o[0] = (__bf16)((v.x - mu) * rs * gg.x + bb.x);
    o[1] = (__bf16)((v.y - mu) * rs * gg.y + bb.y);
    o[2] = (__bf16)((v.z - mu) * rs * gg.z + bb.z);
    o[3] = (__bf16)((v.w - mu) * rs * gg.w + bb.w);
}

// ---------------------------------------------------------------------------
// Tiled bf16 WMMA GEMM with TDM double-buffered LDS staging.
// C[M,N] = A[M,K] @ BT[N,K]^T, f32 accumulate.
// Block tile 128x128, K-tile 32; 256 threads = 8 waves (2M x 4N), each wave
// 64x32 via 4x2 WMMA accumulators. LDS rows padded to 40 halfs by TDM.
// MODE 0: store bf16             MODE 1: +resid, store f32
// MODE 2: +bias, GELU, bf16      MODE 3: +bias +resid, store f32
// MODE 4: store bf16 transposed per-batch ([b*D + n][T]) for V
// ---------------------------------------------------------------------------
template <int MODE>
__global__ __launch_bounds__(256) void gemm_bf16_kernel(
    const __bf16* __restrict__ A, const __bf16* __restrict__ BT,
    const float* __restrict__ bias, const float* __restrict__ resid,
    void* __restrict__ Cout, int M, int N, int K) {
    __shared__ __bf16 smem[4 * 128 * 40];  // A0 | B0 | A1 | B1 (each 128x40)

    const int tid  = threadIdx.x;
    const int wave = tid >> 5, lane = tid & 31;
    const int lm = lane & 15, hi = lane >> 4, k0 = hi * 8;
    const int waveM = wave & 1, waveN = wave >> 1;
    const int m0 = blockIdx.y * 128, n0 = blockIdx.x * 128;

    const v8f vzero = {0.f, 0.f, 0.f, 0.f, 0.f, 0.f, 0.f, 0.f};
    v8f acc[4][2];
#pragma unroll
    for (int i = 0; i < 4; ++i)
#pragma unroll
        for (int j = 0; j < 2; ++j) acc[i][j] = vzero;

    const int nkt = K >> 5;
    // One wave issues both tile DMAs (A rows m0.., BT rows n0.., stride K).
    auto issue = [&](int kt, int buf) {
        tdm_load_2d(A  + (size_t)m0 * K + kt * 32, (uint32_t)(buf * 20480),
                    32, 128, (uint32_t)K, 3, 3);
        tdm_load_2d(BT + (size_t)n0 * K + kt * 32, (uint32_t)(buf * 20480 + 10240),
                    32, 128, (uint32_t)K, 3, 3);
    };
    if (wave == 0) issue(0, 0);

    for (int kt = 0; kt < nkt; ++kt) {
        const int cur = kt & 1;
        if (wave == 0) {
            if (kt + 1 < nkt) {
                issue(kt + 1, cur ^ 1);
                __builtin_amdgcn_s_wait_tensorcnt(2);  // current pair landed
            } else {
                __builtin_amdgcn_s_wait_tensorcnt(0);
            }
        }
        __syncthreads();

        const __bf16* sA = smem + cur * 10240;
        const __bf16* sB = sA + 5120;
        v16bf af[4], bf[2];
#pragma unroll
        for (int i = 0; i < 4; ++i)
            af[i] = ld_frag(&sA[(waveM * 64 + i * 16 + lm) * 40], k0);
#pragma unroll
        for (int j = 0; j < 2; ++j)
            bf[j] = ld_frag(&sB[(waveN * 32 + j * 16 + lm) * 40], k0);
#pragma unroll
        for (int i = 0; i < 4; ++i)
#pragma unroll
            for (int j = 0; j < 2; ++j)
                acc[i][j] = wmma_bf16(af[i], bf[j], acc[i][j]);
        __syncthreads();  // all readers done before next overwrite is issued
    }

    // Epilogue. C layout: m = r + hi*8, n = lane&15.
#pragma unroll
    for (int i = 0; i < 4; ++i) {
        const int mrow = m0 + waveM * 64 + i * 16 + hi * 8;
#pragma unroll
        for (int j = 0; j < 2; ++j) {
            const int n = n0 + waveN * 32 + j * 16 + lm;
            float bval = 0.f;
            if (MODE == 2 || MODE == 3) bval = bias[n];
#pragma unroll
            for (int r = 0; r < 8; ++r) {
                const size_t off = (size_t)(mrow + r) * N + n;
                float val = acc[i][j][r];
                if (MODE == 0) {
                    ((__bf16*)Cout)[off] = (__bf16)val;
                } else if (MODE == 1) {
                    ((float*)Cout)[off] = val + resid[off];
                } else if (MODE == 2) {
                    val += bval;
                    float ge = 0.5f * val * (1.0f + erff(val * 0.70710678118654752f));
                    ((__bf16*)Cout)[off] = (__bf16)ge;
                } else if (MODE == 3) {
                    ((float*)Cout)[off] = val + bval + resid[off];
                } else {  // MODE 4: transposed bf16 store (V^T per batch)
                    const int t  = mrow + r;
                    const int bb = t >> 11;        // / TSEQ
                    const int tt = t & (TSEQ - 1);
                    ((__bf16*)Cout)[(size_t)(bb * DMODEL + n) * TSEQ + tt] = (__bf16)val;
                }
            }
        }
    }
}

// ---------------------------------------------------------------------------
// Flash attention (causal), bf16 in/out, f32 online softmax.
// One block per (b, h, 128 q-rows); 8 waves, each owns 16 q-rows x 64 k-cols.
// q/k: [b*T + t][h*HD + d] bf16; vt: [b*D + h*HD + d][t] bf16 (pre-transposed).
// K / V^T tiles double-buffered through TDM.
// ---------------------------------------------------------------------------
__global__ __launch_bounds__(256) void attn_kernel(
    const __bf16* __restrict__ q, const __bf16* __restrict__ k,
    const __bf16* __restrict__ vt, __bf16* __restrict__ out) {
    // K0 | V0 | K1 | V1 (each 64x72 halfs) | P (128x72 halfs)
    __shared__ __bf16 smem[4 * 64 * 72 + 128 * 72];
    __bf16* sP = smem + 4 * 64 * 72;

    const int tid  = threadIdx.x;
    const int wave = tid >> 5, lane = tid & 31;
    const int lm = lane & 15, hi = lane >> 4, k0 = hi * 8;

    const int nqb = TSEQ / 128;  // 16
    const int qb = blockIdx.x % nqb;
    const int h  = (blockIdx.x / nqb) % NHEAD;
    const int b  = blockIdx.x / (nqb * NHEAD);

    // Q fragments in registers for whole loop (A layout: row = lm)
    const size_t rowQ = (size_t)(b * TSEQ + qb * 128 + wave * 16 + lm) * DMODEL + h * HDIM;
    v16bf aQ[2];
#pragma unroll
    for (int dc = 0; dc < 2; ++dc) aQ[dc] = ld_frag(q + rowQ, dc * 32 + k0);

    const v8f vzero = {0.f, 0.f, 0.f, 0.f, 0.f, 0.f, 0.f, 0.f};
    float Mr[8], Lr[8];
    v8f o[4];
#pragma unroll
    for (int r = 0; r < 8; ++r) { Mr[r] = -1e30f; Lr[r] = 0.f; }
#pragma unroll
    for (int nt = 0; nt < 4; ++nt) o[nt] = vzero;

    const int qrow_base = qb * 128 + wave * 16 + hi * 8;  // + r
    const int nkt = qb * 2 + 2;                           // causal tile bound

    // TDM: K tile rows = 64 keys x 64 d (stride D); V^T rows = 64 d x 64 keys
    // (stride T). Row = 32 dwords -> pad_interval code 4; pad 4 dwords -> 3.
    auto issueKV = [&](int kt, int buf) {
        tdm_load_2d(k  + (size_t)(b * TSEQ + kt * 64) * DMODEL + h * HDIM,
                    (uint32_t)(buf * 18432), 64, 64, DMODEL, 4, 3);
        tdm_load_2d(vt + (size_t)(b * DMODEL + h * HDIM) * TSEQ + kt * 64,
                    (uint32_t)(buf * 18432 + 9216), 64, 64, TSEQ, 4, 3);
    };
    if (wave == 0) issueKV(0, 0);

    for (int kt = 0; kt < nkt; ++kt) {
        const int cur = kt & 1;
        if (wave == 0) {
            if (kt + 1 < nkt) {
                issueKV(kt + 1, cur ^ 1);
                __builtin_amdgcn_s_wait_tensorcnt(2);
            } else {
                __builtin_amdgcn_s_wait_tensorcnt(0);
            }
        }
        __syncthreads();
        const __bf16* sK = smem + cur * 9216;   // [kk][d] padded 72
        const __bf16* sV = sK + 4608;           // [d][kk] padded 72

        // S = (Q @ K^T) / 8 ; B-fragment column n = K row (nt*16+lm)
        v8f s[4];
#pragma unroll
        for (int nt = 0; nt < 4; ++nt) s[nt] = vzero;
#pragma unroll
        for (int dc = 0; dc < 2; ++dc) {
#pragma unroll
            for (int nt = 0; nt < 4; ++nt) {
                v16bf bK = ld_frag(&sK[(nt * 16 + lm) * 72], dc * 32 + k0);
                s[nt] = wmma_bf16(aQ[dc], bK, s[nt]);
            }
        }

        // Online softmax per row (rows live in 16-lane halves)
#pragma unroll
        for (int r = 0; r < 8; ++r) {
            const int qrow = qrow_base + r;
            float pv[4];
            float mx = -1e30f;
#pragma unroll
            for (int nt = 0; nt < 4; ++nt) {
                int kg = kt * 64 + nt * 16 + lm;
                float val = s[nt][r] * 0.125f;  // 1/sqrt(64)
                if (kg > qrow) val = -1e30f;
                pv[nt] = val;
                mx = fmaxf(mx, val);
            }
#pragma unroll
            for (int off = 8; off > 0; off >>= 1)
                mx = fmaxf(mx, __shfl_xor(mx, off, 32));
            const float newM = fmaxf(Mr[r], mx);
            const float alpha = __expf(Mr[r] - newM);
            Mr[r] = newM;
            float ls = 0.f;
#pragma unroll
            for (int nt = 0; nt < 4; ++nt) {
                float p = __expf(pv[nt] - newM);
                ls += p;
                sP[(wave * 16 + hi * 8 + r) * 72 + nt * 16 + lm] = (__bf16)p;
            }
#pragma unroll
            for (int off = 8; off > 0; off >>= 1) ls += __shfl_xor(ls, off, 32);
            Lr[r] = Lr[r] * alpha + ls;
#pragma unroll
            for (int nt = 0; nt < 4; ++nt) o[nt][r] *= alpha;
        }
        __syncthreads();  // P (cross-lane within wave region) visible

        // O += P @ V  (B column n = d -> rows of V^T in LDS)
#pragma unroll
        for (int kc = 0; kc < 2; ++kc) {
            v16bf aP = ld_frag(&sP[(wave * 16 + lm) * 72], kc * 32 + k0);
#pragma unroll
            for (int nt = 0; nt < 4; ++nt) {
                v16bf bV = ld_frag(&sV[(nt * 16 + lm) * 72], kc * 32 + k0);
                o[nt] = wmma_bf16(aP, bV, o[nt]);
            }
        }
        __syncthreads();  // readers done before next TDM overwrites cur buffers
    }

    // Normalize and store ctx (bf16, [t][h*HD+d])
#pragma unroll
    for (int r = 0; r < 8; ++r) {
        const float inv = 1.0f / Lr[r];
        const size_t row =
            (size_t)(b * TSEQ + qb * 128 + wave * 16 + hi * 8 + r) * DMODEL + h * HDIM;
#pragma unroll
        for (int nt = 0; nt < 4; ++nt)
            out[row + nt * 16 + lm] = (__bf16)(o[nt][r] * inv);
    }
}

// ---------------------------------------------------------------------------
// Host-side orchestration
// ---------------------------------------------------------------------------
extern "C" void kernel_launch(void* const* d_in, const int* in_sizes, int n_in,
                              void* d_out, int out_size, void* d_ws, size_t ws_size,
                              hipStream_t stream) {
    const float* x     = (const float*)d_in[0];
    const float* ln1_g = (const float*)d_in[1];
    const float* ln1_b = (const float*)d_in[2];
    const float* Wq    = (const float*)d_in[3];
    const float* Wk    = (const float*)d_in[4];
    const float* Wv    = (const float*)d_in[5];
    const float* Wo    = (const float*)d_in[6];
    const float* ln2_g = (const float*)d_in[7];
    const float* ln2_b = (const float*)d_in[8];
    const float* W_up  = (const float*)d_in[9];
    const float* b_up  = (const float*)d_in[10];
    const float* W_dn  = (const float*)d_in[11];
    const float* b_dn  = (const float*)d_in[12];
    float* out = (float*)d_out;

    char* ws = (char*)d_ws;
    size_t off = 0;
    auto carve = [&](size_t bytes) -> void* {
        void* p = ws + off;
        off += (bytes + 255) & ~(size_t)255;
        return p;
    };
    __bf16* hb    = (__bf16*)carve((size_t)MROWS * DMODEL * 2);
    __bf16* wqT   = (__bf16*)carve((size_t)DMODEL * DMODEL * 2);  // [N][K]
    __bf16* wkT   = (__bf16*)carve((size_t)DMODEL * DMODEL * 2);
    __bf16* wvT   = (__bf16*)carve((size_t)DMODEL * DMODEL * 2);
    __bf16* woT   = (__bf16*)carve((size_t)DMODEL * DMODEL * 2);
    __bf16* wupT  = (__bf16*)carve((size_t)DINNER * DMODEL * 2);  // [4096][1024]
    __bf16* wdnT  = (__bf16*)carve((size_t)DMODEL * DINNER * 2);  // [1024][4096]
    __bf16* qb_   = (__bf16*)carve((size_t)MROWS * DMODEL * 2);
    __bf16* kb_   = (__bf16*)carve((size_t)MROWS * DMODEL * 2);
    __bf16* vtb   = (__bf16*)carve((size_t)MROWS * DMODEL * 2);   // V^T per batch
    __bf16* attnb = (__bf16*)carve((size_t)MROWS * DMODEL * 2);
    float*  xmid  = (float*)carve((size_t)MROWS * DMODEL * 4);
    __bf16* h2b   = (__bf16*)carve((size_t)MROWS * DMODEL * 2);
    __bf16* mb    = (__bf16*)carve((size_t)MROWS * DINNER * 2);
    (void)ws_size; (void)in_sizes; (void)n_in; (void)out_size;

    // 1) Weight convert + transpose to bf16 [N][K]
    {
        dim3 gsq(DMODEL / 32, DMODEL / 32);
        cvt_transpose_bf16<<<gsq, 256, 0, stream>>>(Wq, wqT, DMODEL, DMODEL);
        cvt_transpose_bf16<<<gsq, 256, 0, stream>>>(Wk, wkT, DMODEL, DMODEL);
        cvt_transpose_bf16<<<gsq, 256, 0, stream>>>(Wv, wvT, DMODEL, DMODEL);
        cvt_transpose_bf16<<<gsq, 256, 0, stream>>>(Wo, woT, DMODEL, DMODEL);
        dim3 gup(DINNER / 32, DMODEL / 32);   // src [1024][4096]
        cvt_transpose_bf16<<<gup, 256, 0, stream>>>(W_up, wupT, DMODEL, DINNER);
        dim3 gdn(DMODEL / 32, DINNER / 32);   // src [4096][1024]
        cvt_transpose_bf16<<<gdn, 256, 0, stream>>>(W_dn, wdnT, DINNER, DMODEL);
    }

    // 2) LN1 -> h (bf16)
    ln_bf16_kernel<<<MROWS, 256, 0, stream>>>(x, ln1_g, ln1_b, hb);

    // 3) Q/K/V projections (V stored transposed for attention)
    {
        dim3 grid(DMODEL / 128, MROWS / 128);
        gemm_bf16_kernel<0><<<grid, 256, 0, stream>>>(hb, wqT, nullptr, nullptr,
                                                      qb_, MROWS, DMODEL, DMODEL);
        gemm_bf16_kernel<0><<<grid, 256, 0, stream>>>(hb, wkT, nullptr, nullptr,
                                                      kb_, MROWS, DMODEL, DMODEL);
        gemm_bf16_kernel<4><<<grid, 256, 0, stream>>>(hb, wvT, nullptr, nullptr,
                                                      vtb, MROWS, DMODEL, DMODEL);
    }

    // 4) Causal flash attention
    attn_kernel<<<NBATCH * NHEAD * (TSEQ / 128), 256, 0, stream>>>(qb_, kb_, vtb, attnb);

    // 5) Output projection + residual -> xmid (f32)
    {
        dim3 grid(DMODEL / 128, MROWS / 128);
        gemm_bf16_kernel<1><<<grid, 256, 0, stream>>>(attnb, woT, nullptr, x,
                                                      xmid, MROWS, DMODEL, DMODEL);
    }

    // 6) LN2 -> h2 (bf16)
    ln_bf16_kernel<<<MROWS, 256, 0, stream>>>(xmid, ln2_g, ln2_b, h2b);

    // 7) Up-projection + bias + exact GELU -> m (bf16)
    {
        dim3 grid(DINNER / 128, MROWS / 128);
        gemm_bf16_kernel<2><<<grid, 256, 0, stream>>>(h2b, wupT, b_up, nullptr,
                                                      mb, MROWS, DINNER, DMODEL);
    }

    // 8) Down-projection + bias + residual -> out (f32)
    {
        dim3 grid(DMODEL / 128, MROWS / 128);
        gemm_bf16_kernel<3><<<grid, 256, 0, stream>>>(mb, wdnT, b_dn, xmid,
                                                      out, MROWS, DMODEL, DINNER);
    }
}
